// XlAttention_22308060135837
// MI455X (gfx1250) — matrix operand order repaired and checked by reading, not compiled, not run on hardware
//
#include <hip/hip_runtime.h>
#include <cstdint>

// ---------------------------------------------------------------------------
// Transformer-XL relative attention for MI455X (gfx1250, wave32, WMMA bf16).
// Compute-bound (~100 GFLOP vs ~150MB traffic @23.3TB/s) -> all matmuls on
// v_wmma_f32_16x16x32_bf16; staging uses CDNA5 TDM (tensor_load_to_lds) /
// global_load_async_to_lds_b128 instead of VGPR round-trips.
// ---------------------------------------------------------------------------

typedef __attribute__((ext_vector_type(16))) __bf16 v16bf;
typedef __attribute__((ext_vector_type(8)))  __bf16 v8bf;
typedef __attribute__((ext_vector_type(4)))  __bf16 v4bf;
typedef __attribute__((ext_vector_type(8)))  float  v8f;

#define D_MODEL 1024
#define N_HEAD  16
#define D_HEAD  64
#define BATCH   4
#define QLEN    1024
#define MLEN    1024
#define KLEN    2048   // == RLEN

#define BQ 32          // query tile rows per attention block
#define BK 64          // key tile cols per attention block

#if defined(__has_builtin)
#if __has_builtin(__builtin_amdgcn_tensor_load_to_lds) && \
    __has_builtin(__builtin_amdgcn_s_wait_tensorcnt)
#define USE_TDM 1
#endif
#endif
#ifndef USE_TDM
#define USE_TDM 0
#endif

static __device__ __forceinline__ v16bf mk16(v8bf lo, v8bf hi) {
  v16bf r;
#pragma unroll
  for (int i = 0; i < 8; ++i) { r[i] = lo[i]; r[i + 8] = hi[i]; }
  return r;
}
static __device__ __forceinline__ v8f vzero8() {
  v8f r;
#pragma unroll
  for (int i = 0; i < 8; ++i) r[i] = 0.f;
  return r;
}

// ---- CDNA5 async copy: per-lane 16B global -> LDS (ASYNCcnt) ---------------
static __device__ __forceinline__ void async_ld_b128(void* lds, const void* g) {
  uint32_t l = (uint32_t)(uintptr_t)lds;  // flat LDS addr low 32 bits = LDS offset
  asm volatile("global_load_async_to_lds_b128 %0, %1, off"
               :: "v"(l), "v"(g) : "memory");
}
static __device__ __forceinline__ void wait_async0() {
  asm volatile("s_wait_asynccnt 0x0" ::: "memory");
}

#if USE_TDM
// ---- CDNA5 Tensor Data Mover: 2D tile (16-bit elements) global -> LDS ------
// D# per ISA 8.3/8.4: group0 = {count, lds_addr, global_addr, type=2},
// group1 = {data_size=2B, tensor dims == tile dims (no OOB), tile dims,
// tensor_dim0_stride}. Groups 2/3 zero (2D tensor). This toolchain exposes
// the 6-arg builtin: (u32x4, i32x8, i32x4, i32x4, i32x8, i32 cpol).
typedef __attribute__((ext_vector_type(4))) unsigned int tdm_v4u;
typedef __attribute__((ext_vector_type(8))) int          tdm_v8i;
typedef __attribute__((ext_vector_type(4))) int          tdm_v4i;

static __device__ __forceinline__ void tdm_load_2d_bf16(
    void* lds, const void* gptr, uint32_t tile_d0, uint32_t tile_d1,
    uint32_t stride_elems) {
  uint64_t ga = (uint64_t)(uintptr_t)gptr;
  tdm_v4u g0;
  g0[0] = 1u;                                         // count=1 (valid user D#)
  g0[1] = (uint32_t)(uintptr_t)lds;                   // lds_addr (bytes)
  g0[2] = (uint32_t)ga;                               // global_addr[31:0]
  g0[3] = (uint32_t)((ga >> 32) & 0x01FFFFFFu) | (2u << 30); // addr[56:32]|type=2
  tdm_v8i g1;
  g1[0] = (int)(1u << 16);                            // data_size=1 -> 2 bytes
  g1[1] = (int)((tile_d0 & 0xFFFFu) << 16);           // tensor_dim0[15:0]
  g1[2] = (int)(((tile_d0 >> 16) & 0xFFFFu) |
                ((tile_d1 & 0xFFFFu) << 16));         // t_dim0 hi | t_dim1 lo
  g1[3] = (int)(((tile_d1 >> 16) & 0xFFFFu) |
                ((tile_d0 & 0xFFFFu) << 16));         // t_dim1 hi | tile_dim0
  g1[4] = (int)(tile_d1 & 0xFFFFu);                   // tile_dim1 (tile_dim2=0)
  g1[5] = (int)stride_elems;                          // tensor_dim0_stride lo32
  g1[6] = 0;                                          // stride hi | dim1_stride
  g1[7] = 0;
  tdm_v4i gz4 = {0, 0, 0, 0};
  tdm_v8i gz8 = {0, 0, 0, 0, 0, 0, 0, 0};
  __builtin_amdgcn_tensor_load_to_lds(g0, g1, gz4, gz4, gz8, 0);
}
#endif

// ---------------------------------------------------------------------------
// Elementwise helpers
// ---------------------------------------------------------------------------
__global__ void cvt_f32_bf16(const float* __restrict__ x, __bf16* __restrict__ y,
                             long n) {
  long i = (long)blockIdx.x * blockDim.x + threadIdx.x;
  long stride = (long)gridDim.x * blockDim.x;
  for (; i < n; i += stride) y[i] = (__bf16)x[i];
}

__global__ void qprep_kernel(const float* __restrict__ q,
                             const float* __restrict__ bw,
                             const float* __restrict__ br,
                             __bf16* __restrict__ qw, __bf16* __restrict__ qr,
                             long n) {
  long i = (long)blockIdx.x * blockDim.x + threadIdx.x;
  long stride = (long)gridDim.x * blockDim.x;
  for (; i < n; i += stride) {
    int c = (int)(i & 1023);               // h*64 + d
    float v = q[i];
    qw[i] = (__bf16)(v + bw[c]);
    qr[i] = (__bf16)(v + br[c]);
  }
}

__global__ void kvsplit_kernel(const float* __restrict__ kv,
                               __bf16* __restrict__ k, __bf16* __restrict__ v,
                               long n) {
  long i = (long)blockIdx.x * blockDim.x + threadIdx.x;
  long stride = (long)gridDim.x * blockDim.x;
  for (; i < n; i += stride) {
    long row = i >> 11;
    int c = (int)(i & 2047);
    float x = kv[i];
    if (c < 1024) k[row * 1024 + c] = (__bf16)x;
    else          v[row * 1024 + (c - 1024)] = (__bf16)x;
  }
}

// ---------------------------------------------------------------------------
// Tiled bf16 GEMM: C[M,N] f32 = A[M,K] @ B[K,N], row-major, M,N mult of 128,
// K mult of 32. 256 thr = 8 waves (4x2), each wave owns 32x64 of C.
// A tile: TDM 2D DMA (or async b128). B tile: async b128 raw stage, then
// LDS->LDS transpose so B fragments (column-per-lane) are contiguous reads.
// ---------------------------------------------------------------------------
__global__ __launch_bounds__(256) void gemm_bf16_wmma(
    const __bf16* __restrict__ A, const __bf16* __restrict__ B,
    float* __restrict__ C, int M, int N, int K) {
  __shared__ __bf16 Alds[128][32];
  __shared__ __bf16 Braw[32][128];   // raw row-major B tile
  __shared__ __bf16 Blds[128][32];   // transposed: Blds[n][k]
  const int m0 = blockIdx.y * 128, n0 = blockIdx.x * 128;
  const int tid = threadIdx.x, wave = tid >> 5, lane = tid & 31;
  const int lrow = lane & 15, half = lane >> 4, koff = half * 8;
  const int wr = wave >> 1, wc = wave & 1;
  v8f acc[2][4];
#pragma unroll
  for (int i = 0; i < 2; ++i)
#pragma unroll
    for (int j = 0; j < 4; ++j) acc[i][j] = vzero8();

  for (int k0 = 0; k0 < K; k0 += 32) {
    __syncthreads();
    // ---- A tile 128x32: one TDM descriptor (tile_d0=32, 128 rows, stride K)
#if USE_TDM
    if (wave == 0)
      tdm_load_2d_bf16(&Alds[0][0], &A[(size_t)m0 * K + k0], 32u, 128u,
                       (uint32_t)K);
#else
#pragma unroll
    for (int it = 0; it < 2; ++it) {
      int c = tid * 2 + it;              // 512 x 16B chunks
      int rr = c >> 2, cc = (c & 3) * 8;
      async_ld_b128(&Alds[rr][cc], &A[(size_t)(m0 + rr) * K + k0 + cc]);
    }
#endif
    // ---- B tile 32x128 raw: one async b128 chunk per thread
    {
      int rr = tid >> 3;                 // 0..31
      int cc = (tid & 7) * 16;           // 0..112
      async_ld_b128(&Braw[rr][cc], &B[(size_t)(k0 + rr) * N + n0 + cc]);
    }
    if (k0 + 32 < K) {  // gfx1250 global_prefetch_b8 for next K slab
      __builtin_prefetch(&A[(size_t)(m0 + (tid & 127)) * K + k0 + 32], 0, 1);
      __builtin_prefetch(&B[(size_t)(k0 + 32 + (tid & 31)) * N + n0], 0, 1);
    }
    wait_async0();
#if USE_TDM
    if (wave == 0) __builtin_amdgcn_s_wait_tensorcnt(0);
#endif
    __syncthreads();
    // ---- transpose Braw -> Blds (vector LDS reads, scalar b16 stores)
#pragma unroll
    for (int e = tid; e < 512; e += 256) {
      int kk = e >> 4;                   // 0..31
      int nb = (e & 15) * 8;             // 0..120
      v8bf row = *(const v8bf*)&Braw[kk][nb];
#pragma unroll
      for (int t = 0; t < 8; ++t) Blds[nb + t][kk] = row[t];
    }
    __syncthreads();
#pragma unroll
    for (int i = 0; i < 2; ++i) {
      int ar = 32 * wr + 16 * i + lrow;
      v16bf a = mk16(*(const v8bf*)&Alds[ar][16 * half],
                     *(const v8bf*)&Alds[ar][16 * half + 8]);
#pragma unroll
      for (int j = 0; j < 4; ++j) {
        int bc = 64 * wc + 16 * j + lrow;
        v16bf bb = mk16(*(const v8bf*)&Blds[bc][16 * half],
                        *(const v8bf*)&Blds[bc][16 * half + 8]);
        acc[i][j] = __builtin_amdgcn_wmma_f32_16x16x32_bf16(
            false, a, false, bb, (short)0, acc[i][j], false, false);
      }
    }
  }
#pragma unroll
  for (int i = 0; i < 2; ++i)
#pragma unroll
    for (int j = 0; j < 4; ++j)
#pragma unroll
      for (int r = 0; r < 8; ++r) {
        int row = m0 + 32 * wr + 16 * i + r + 8 * half;
        int col = n0 + 64 * wc + 16 * j + lrow;
        C[(size_t)row * N + col] = acc[i][j][r];
      }
}

// ---------------------------------------------------------------------------
// Fused XL attention (flash-style). Block = (32-query tile, head, batch).
// AC via WMMA from global; BD via WMMA over rel-shift band of rk rows with
// the diagonal gathered through LDS; V tile staged by TDM/async then
// transposed in LDS; online softmax; PV accumulators live in registers.
// ---------------------------------------------------------------------------
__global__ __launch_bounds__(256) void xl_attn_kernel(
    const __bf16* __restrict__ qw, const __bf16* __restrict__ qr,
    const __bf16* __restrict__ kb, const __bf16* __restrict__ vb,
    const __bf16* __restrict__ rkb, float* __restrict__ attn) {
  __shared__ float acS[BQ][BK + 4];
  __shared__ float bdS[BQ][BQ + BK + 8];
  __shared__ __bf16 Vraw[BK][D_HEAD];     // raw V tile (rows = keys)
  __shared__ __bf16 Vt[D_HEAD][BK + 8];   // transposed V tile
  __shared__ __bf16 P[BQ][BK + 8];
  __shared__ float rowscale[BQ];
  __shared__ float rowinv[BQ];

  const int qt = blockIdx.x, h = blockIdx.y, b = blockIdx.z;
  const int i0 = qt * BQ;
  const int tid = threadIdx.x, wave = tid >> 5, lane = tid & 31;
  const int lrow = lane & 15, half = lane >> 4, koff = half * 8;
  const int wr = wave >> 2, wc = wave & 3;   // 2x4 wave grid over 32x64 tile

  const __bf16* qwBase = qw + (size_t)(b * QLEN) * 1024 + h * D_HEAD;
  const __bf16* qrBase = qr + (size_t)(b * QLEN) * 1024 + h * D_HEAD;
  const __bf16* kBase  = kb + (size_t)(b * KLEN) * 1024 + h * D_HEAD;
  const __bf16* vBase  = vb + (size_t)(b * KLEN) * 1024 + h * D_HEAD;

  // AC A-fragments: fixed query rows for this wave, load once.
  const __bf16* aptr = qwBase + (size_t)(i0 + 16 * wr + lrow) * 1024;
  v16bf aAC[2];
#pragma unroll
  for (int s = 0; s < 2; ++s)
    aAC[s] = mk16(*(const v8bf*)(aptr + 32 * s + 16 * half),
                  *(const v8bf*)(aptr + 32 * s + 16 * half + 8));

  v8f Oacc = vzero8();
  float m_i = -3.0e38f, l_i = 0.f;   // per-row state (threads 0..31)

  const int nkb = (i0 + BQ + MLEN + BK - 1) / BK;
  for (int jb = 0; jb < nkb; ++jb) {
    const int j0 = jb * BK;
    __syncthreads();   // prev PV done with P/Vt; prev softmax done with acS/bdS

    // ---- V tile 64x64 into Vraw: TDM 2D DMA (or per-lane async b128)
#if USE_TDM
    if (wave == 0)
      tdm_load_2d_bf16(&Vraw[0][0], vBase + (size_t)j0 * 1024, (uint32_t)D_HEAD,
                       (uint32_t)BK, 1024u);
#else
#pragma unroll
    for (int it = 0; it < 2; ++it) {
      int c = tid + it * 256;            // 512 x 16B chunks
      int rr = c >> 3, cc = (c & 7) * 8;
      async_ld_b128(&Vraw[rr][cc], vBase + (size_t)(j0 + rr) * 1024 + cc);
    }
#endif

    // ---- AC tile: this wave's 16x16 at (16*wr, 16*wc)
    {
      const __bf16* bptr = kBase + (size_t)(j0 + 16 * wc + lrow) * 1024;
      v8f c = vzero8();
#pragma unroll
      for (int s = 0; s < 2; ++s) {
        v16bf bbf = mk16(*(const v8bf*)(bptr + 32 * s + 16 * half),
                         *(const v8bf*)(bptr + 32 * s + 16 * half + 8));
        c = __builtin_amdgcn_wmma_f32_16x16x32_bf16(
            false, aAC[s], false, bbf, (short)0, c, false, false);
      }
#pragma unroll
      for (int r = 0; r < 8; ++r)
        acS[16 * wr + r + 8 * half][16 * wc + lrow] = c[r];
    }

    // ---- BD band: 2 x 6 = 12 tiles over rk rows [rbase, rbase+96)
    {
      const int rbase = QLEN - BQ - i0 + j0;   // >= 0 always
#pragma unroll
      for (int tt = 0; tt < 2; ++tt) {
        const int t = wave + tt * 8;
        if (t < 12) {
          const int tr = t / 6, tc = t % 6;
          const __bf16* ar = qrBase + (size_t)(i0 + 16 * tr + lrow) * 1024;
          int jr = rbase + 16 * tc + lrow;
          if (jr > KLEN - 1) jr = KLEN - 1;    // OOB band entries are masked
          const __bf16* bp = rkb + (size_t)jr * 1024 + h * D_HEAD;
          v8f c = vzero8();
#pragma unroll
          for (int s = 0; s < 2; ++s) {
            v16bf a = mk16(*(const v8bf*)(ar + 32 * s + 16 * half),
                           *(const v8bf*)(ar + 32 * s + 16 * half + 8));
            v16bf bbf = mk16(*(const v8bf*)(bp + 32 * s + 16 * half),
                             *(const v8bf*)(bp + 32 * s + 16 * half + 8));
            c = __builtin_amdgcn_wmma_f32_16x16x32_bf16(
                false, a, false, bbf, (short)0, c, false, false);
          }
#pragma unroll
          for (int r = 0; r < 8; ++r)
            bdS[16 * tr + r + 8 * half][16 * tc + lrow] = c[r];
        }
      }
    }

#if USE_TDM
    if (wave == 0) __builtin_amdgcn_s_wait_tensorcnt(0);
#else
    wait_async0();
#endif
    __syncthreads();

    // ---- online softmax: thread i < 32 owns query row i0+i
    if (tid < BQ) {
      const int i = tid;
      const int ig = i0 + i;
      float mx = m_i;
      for (int j = 0; j < BK; ++j) {
        // rel-shift gather: column j + (BQ-1) - i of the BD band
        float s = (acS[i][j] + bdS[i][j + (BQ - 1) - i]) * 0.125f;
        if (j0 + j > ig + MLEN) s = -1.0e30f;   // causal-with-memory mask
        acS[i][j] = s;
        mx = fmaxf(mx, s);
      }
      float alpha = __expf(m_i - mx);
      rowscale[i] = alpha;
      float sum = 0.f;
      for (int j = 0; j < BK; ++j) {
        float pv = __expf(acS[i][j] - mx);
        sum += pv;
        P[i][j] = (__bf16)pv;
      }
      l_i = l_i * alpha + sum;
      m_i = mx;
    }
    // ---- transpose Vraw -> Vt (all threads; vector reads, b16 stores)
#pragma unroll
    for (int e = tid; e < 512; e += 256) {
      int jj = e >> 3;
      int db = (e & 7) * 8;
      v8bf row = *(const v8bf*)&Vraw[jj][db];
#pragma unroll
      for (int t = 0; t < 8; ++t) Vt[db + t][jj] = row[t];
    }
    __syncthreads();

    // ---- rescale running output, then PV accumulate (64 keys = 2 k-steps)
#pragma unroll
    for (int r = 0; r < 8; ++r)
      Oacc[r] *= rowscale[16 * wr + r + 8 * half];
#pragma unroll
    for (int s = 0; s < 2; ++s) {
      v16bf a = mk16(*(const v8bf*)&P[16 * wr + lrow][32 * s + 16 * half],
                     *(const v8bf*)&P[16 * wr + lrow][32 * s + 16 * half + 8]);
      v16bf bbf = mk16(*(const v8bf*)&Vt[16 * wc + lrow][32 * s + 16 * half],
                       *(const v8bf*)&Vt[16 * wc + lrow][32 * s + 16 * half + 8]);
      Oacc = __builtin_amdgcn_wmma_f32_16x16x32_bf16(
          false, a, false, bbf, (short)0, Oacc, false, false);
    }
  }

  if (tid < BQ) rowinv[tid] = 1.0f / l_i;
  __syncthreads();
#pragma unroll
  for (int r = 0; r < 8; ++r) {
    int row = 16 * wr + r + 8 * half;
    attn[(size_t)(b * QLEN + i0 + row) * 1024 + h * D_HEAD + 16 * wc + lrow] =
        Oacc[r] * rowinv[row];
  }
}

// ---------------------------------------------------------------------------
// Host orchestration
// ---------------------------------------------------------------------------
extern "C" void kernel_launch(void* const* d_in, const int* in_sizes, int n_in,
                              void* d_out, int out_size, void* d_ws,
                              size_t ws_size, hipStream_t stream) {
  (void)in_sizes; (void)n_in; (void)out_size; (void)ws_size;
  const float* w      = (const float*)d_in[0];  // [4,2048,1024]
  const float* r      = (const float*)d_in[1];  // [1,2048,1024]
  const float* Wq     = (const float*)d_in[2];  // [1024,1024]
  const float* Wkv    = (const float*)d_in[3];  // [1024,2048]
  const float* Wr     = (const float*)d_in[4];  // [1024,1024]
  const float* Wo     = (const float*)d_in[5];  // [1024,1024]
  const float* bias_w = (const float*)d_in[6];  // [16,64]
  const float* bias_r = (const float*)d_in[7];  // [16,64]
  // d_in[8] attn_mask (computed analytically), d_in[9] qlen (compile-time)

  char* p = (char*)d_ws;
  auto alloc = [&](size_t bytes) {
    void* q = (void*)p;
    p += (bytes + 255) & ~(size_t)255;
    return q;
  };
  const size_t MQ = (size_t)BATCH * QLEN;   // 4096
  const size_t MK = (size_t)BATCH * KLEN;   // 8192

  __bf16* wb   = (__bf16*)alloc(MK * 1024 * 2);
  __bf16* rb   = (__bf16*)alloc((size_t)KLEN * 1024 * 2);
  __bf16* Wqb  = (__bf16*)alloc((size_t)1024 * 1024 * 2);
  __bf16* Wkvb = (__bf16*)alloc((size_t)1024 * 2048 * 2);
  __bf16* Wrb  = (__bf16*)alloc((size_t)1024 * 1024 * 2);
  __bf16* Wob  = (__bf16*)alloc((size_t)1024 * 1024 * 2);
  float*  qf   = (float*) alloc(MQ * 1024 * 4);
  float*  kvf  = (float*) alloc(MK * 2048 * 4);
  float*  rkf  = (float*) alloc((size_t)KLEN * 1024 * 4);
  __bf16* qwb  = (__bf16*)alloc(MQ * 1024 * 2);
  __bf16* qrb  = (__bf16*)alloc(MQ * 1024 * 2);
  __bf16* kbb  = (__bf16*)alloc(MK * 1024 * 2);
  __bf16* vbb  = (__bf16*)alloc(MK * 1024 * 2);
  __bf16* rkb  = (__bf16*)alloc((size_t)KLEN * 1024 * 2);
  float*  attf = (float*) alloc(MQ * 1024 * 4);
  __bf16* attb = (__bf16*)alloc(MQ * 1024 * 2);

  auto cvt = [&](const float* src, __bf16* dst, long n) {
    long blocks = (n + 255) / 256; if (blocks > 4096) blocks = 4096;
    cvt_f32_bf16<<<dim3((unsigned)blocks), dim3(256), 0, stream>>>(src, dst, n);
  };
  cvt(w,   wb,   (long)MK * 1024);
  cvt(r,   rb,   (long)KLEN * 1024);
  cvt(Wq,  Wqb,  1024L * 1024);
  cvt(Wkv, Wkvb, 1024L * 2048);
  cvt(Wr,  Wrb,  1024L * 1024);
  cvt(Wo,  Wob,  1024L * 1024);

  // q projection: per batch (last QLEN rows of w are non-contiguous across b)
  for (int b = 0; b < BATCH; ++b) {
    gemm_bf16_wmma<<<dim3(1024 / 128, 1024 / 128), 256, 0, stream>>>(
        wb + (size_t)(b * KLEN + MLEN) * 1024, Wqb,
        qf + (size_t)b * QLEN * 1024, 1024, 1024, 1024);
  }
  // kv projection: [8192,1024] @ [1024,2048]
  gemm_bf16_wmma<<<dim3(2048 / 128, 8192 / 128), 256, 0, stream>>>(
      wb, Wkvb, kvf, 8192, 2048, 1024);
  // rk projection: [2048,1024] @ [1024,1024]
  gemm_bf16_wmma<<<dim3(1024 / 128, 2048 / 128), 256, 0, stream>>>(
      rb, Wrb, rkf, 2048, 1024, 1024);

  {
    long n = (long)MQ * 1024;
    qprep_kernel<<<dim3(4096), dim3(256), 0, stream>>>(qf, bias_w, bias_r,
                                                       qwb, qrb, n);
  }
  {
    long n = (long)MK * 2048;
    kvsplit_kernel<<<dim3(4096), dim3(256), 0, stream>>>(kvf, kbb, vbb, n);
  }
  cvt(rkf, rkb, (long)KLEN * 1024);

  // fused attention: grid (q-tiles, heads, batch)
  xl_attn_kernel<<<dim3(QLEN / BQ, N_HEAD, BATCH), 256, 0, stream>>>(
      qwb, qrb, kbb, vbb, rkb, attf);

  cvt(attf, attb, (long)MQ * 1024);

  // output projection -> d_out (f32)
  gemm_bf16_wmma<<<dim3(1024 / 128, 4096 / 128), 256, 0, stream>>>(
      attb, Wob, (float*)d_out, 4096, 1024, 1024);
}
